// InterNet_6562710028555
// MI455X (gfx1250) — matrix operand order repaired and checked by reading, not compiled, not run on hardware
//
#include <hip/hip_runtime.h>

// ---------------------------------------------------------------------------
// InterNet block for MI455X (gfx1250, wave32, WMMA).
// All 3x3 convs run as implicit-GEMM via v_wmma_f32_16x16x32_f16 (f32 accum).
// Pairwise relation conv is factored through conv linearity:
//   conv(concat(xi,xj), w_rel) = convA(xi) + convB(xj)
// reducing 15360 (128->64) convs to 2048 (64->64) convs + cheap aggregation.
// Input staging uses the gfx1250 async global->LDS path (ASYNCcnt).
// ---------------------------------------------------------------------------

typedef __attribute__((ext_vector_type(16))) _Float16 v16h;
typedef __attribute__((ext_vector_type(8)))  float    v8f;

union FragU { v16h h; uint4 q[2]; };

// ---------------------------------------------------------------------------
// Generic 3x3 SAME conv, CIN->COUT, 5x5 images, one wave per sample.
// flags: 1 = relu, 2 = add addsrc BEFORE relu, 4 = add addsrc AFTER relu.
// ---------------------------------------------------------------------------
template<int CIN, int COUT>
__global__ __launch_bounds__(32)
void conv3x3_wmma_kernel(const float* __restrict__ xin,        // [S][CIN][25]
                         const _Float16* __restrict__ wh,      // [COUT][CIN*9] f16 row-major
                         const float* __restrict__ bias,       // [COUT] or null
                         const float* __restrict__ addsrc,     // [S][COUT][25] or null
                         float* __restrict__ out,              // [S][COUT][25]
                         int flags)
{
    constexpr int KPAD  = CIN * 9;    // 576 or 288, multiple of 32
    constexpr int KSTEP = KPAD / 32;
    constexpr int MT    = COUT / 16;
    constexpr int XBYTES = CIN * 25 * 4;   // fp32 sample size in bytes

    __shared__ __attribute__((aligned(16))) _Float16 xpad[CIN * 49];   // 7x7 zero-padded f16
    __shared__ __attribute__((aligned(16))) _Float16 bmat[32 * KPAD];  // [pixel][K] im2col

    const int lane = threadIdx.x;
    const int s    = blockIdx.x;
    const float* xsamp = xin + (size_t)s * CIN * 25;

    // --- Stage fp32 sample into LDS via async global->LDS DMA (ASYNCcnt). ---
    // Staging aliases the first XBYTES of bmat (bmat written only later).
    float* xstage = (float*)bmat;
    {
        char* sbase = (char*)xstage;
        for (int ofs = lane * 16; ofs < XBYTES; ofs += 32 * 16) {
            unsigned ldsoff = (unsigned)(uintptr_t)(sbase + ofs);   // low 32b = LDS offset
            const char* gptr = (const char*)xsamp + ofs;
            asm volatile("global_load_async_to_lds_b128 %0, %1, off"
                         :: "v"(ldsoff), "v"(gptr) : "memory");
        }
        asm volatile("s_wait_asynccnt 0x0" ::: "memory");
    }
    __syncthreads();

    // zero-fill pad region, then convert staged fp32 -> f16 padded 7x7
    for (int idx = lane; idx < CIN * 49; idx += 32) xpad[idx] = (_Float16)0.0f;
    __syncthreads();
    for (int idx = lane; idx < CIN * 25; idx += 32) {
        int ic = idx / 25, p = idx % 25;
        xpad[ic * 49 + (p / 5 + 1) * 7 + (p % 5 + 1)] = (_Float16)xstage[idx];
    }
    __syncthreads();

    // im2col: lane = pixel column; K contiguous per pixel (pixels 25..31 zero)
    {
        int pix = lane;
        _Float16* brow = &bmat[(size_t)pix * KPAD];
        if (pix < 25) {
            int py = pix / 5, px = pix % 5;
            for (int ic = 0; ic < CIN; ++ic) {
                const _Float16* xr = &xpad[ic * 49];
                int kb = ic * 9;
                #pragma unroll
                for (int ky = 0; ky < 3; ++ky)
                    #pragma unroll
                    for (int kx = 0; kx < 3; ++kx)
                        brow[kb + ky * 3 + kx] = xr[(py + ky) * 7 + (px + kx)];
            }
        } else {
            for (int k2 = 0; k2 < KPAD; ++k2) brow[k2] = (_Float16)0.0f;
        }
    }
    __syncthreads();

    const int m16 = lane & 15;
    const int g   = lane >> 4;   // half-wave select

    v8f acc[MT][2];
    #pragma unroll
    for (int mt = 0; mt < MT; ++mt)
        #pragma unroll
        for (int nt = 0; nt < 2; ++nt)
            acc[mt][nt] = (v8f)(0.0f);

    for (int kb = 0; kb < KSTEP; ++kb) {
        const int k0 = kb * 32;

        // B fragments (32K x 16N f16): lane=col n, elems = K g*16..g*16+15 contiguous
        FragU fb[2];
        #pragma unroll
        for (int nt = 0; nt < 2; ++nt) {
            const _Float16* bp = &bmat[(size_t)(nt * 16 + m16) * KPAD + k0 + g * 16];
            fb[nt].q[0] = *(const uint4*)(bp);
            fb[nt].q[1] = *(const uint4*)(bp + 8);
        }
        // A fragments (16M x 32K f16): lane=row m; elems 0..15 -> K g*16..g*16+15
        // contiguous (matches B fragment k order)
        FragU fa[MT];
        #pragma unroll
        for (int mt = 0; mt < MT; ++mt) {
            const _Float16* ap = wh + (size_t)(mt * 16 + m16) * KPAD + k0;
            fa[mt].q[0] = *(const uint4*)(ap + g * 16);
            fa[mt].q[1] = *(const uint4*)(ap + g * 16 + 8);
        }
        if (kb + 1 < KSTEP)
            __builtin_prefetch(wh + (size_t)m16 * KPAD + k0 + 32, 0, 1);

        #pragma unroll
        for (int mt = 0; mt < MT; ++mt)
            #pragma unroll
            for (int nt = 0; nt < 2; ++nt)
                acc[mt][nt] = __builtin_amdgcn_wmma_f32_16x16x32_f16(
                    false, fa[mt].h, false, fb[nt].h,
                    (short)0, acc[mt][nt], false, false);
    }

    // Store: f32 C/D layout: VGPR r -> M = r + 8*g, N = lane&15
    float* outp = out + (size_t)s * COUT * 25;
    const float* addp = addsrc ? (addsrc + (size_t)s * COUT * 25) : nullptr;
    #pragma unroll
    for (int mt = 0; mt < MT; ++mt) {
        #pragma unroll
        for (int nt = 0; nt < 2; ++nt) {
            int pix = nt * 16 + m16;
            if (pix < 25) {
                #pragma unroll
                for (int r = 0; r < 8; ++r) {
                    int oc = mt * 16 + g * 8 + r;
                    float val = acc[mt][nt][r];
                    if (bias) val += bias[oc];
                    size_t oidx = (size_t)oc * 25 + pix;
                    if ((flags & 2) && addp) val += addp[oidx];
                    if (flags & 1) val = fmaxf(val, 0.0f);
                    if ((flags & 4) && addp) val += addp[oidx];
                    outp[oidx] = val;
                }
            }
        }
    }
}

// ---------------------------------------------------------------------------
// f32 -> f16 weight repack: dst[oc][ic*9+t] = src[oc][icoff+ic][t]
// ---------------------------------------------------------------------------
__global__ void cvt_w_kernel(const float* __restrict__ src, _Float16* __restrict__ dst,
                             int cout, int cintot, int icoff, int cinsel)
{
    int n = cout * cinsel * 9;
    for (int idx = blockIdx.x * blockDim.x + threadIdx.x; idx < n;
         idx += gridDim.x * blockDim.x) {
        int oc = idx / (cinsel * 9);
        int r  = idx % (cinsel * 9);
        int ic = r / 9, t = r % 9;
        dst[idx] = (_Float16)src[((size_t)oc * cintot + icoff + ic) * 9 + t];
    }
}

// ---------------------------------------------------------------------------
// 1x1 conv D->DS (tiny): s2d[s][oc][pix] = b + sum_ic w[oc][ic]*x[s][ic][pix]
// ---------------------------------------------------------------------------
__global__ void s2d_kernel(const float* __restrict__ x, const float* __restrict__ w_attn,
                           const float* __restrict__ b_attn, float* __restrict__ s2d)
{
    int idx = blockIdx.x * blockDim.x + threadIdx.x;
    if (idx >= 1024 * 32 * 25) return;
    int samp = idx / 800, rem = idx % 800;
    int oc = rem / 25, pix = rem % 25;
    const float* xp = x + (size_t)samp * 64 * 25;
    float acc = b_attn[oc];
    #pragma unroll 8
    for (int ic = 0; ic < 64; ++ic)
        acc += w_attn[oc * 64 + ic] * xp[ic * 25 + pix];
    s2d[idx] = acc;
}

// ---------------------------------------------------------------------------
// Attention: attn = q.k^T / sqrt(800), row softmax -> self/rel weights.
// One block per batch; thread (i,j) does one 800-dot.
// ---------------------------------------------------------------------------
__global__ __launch_bounds__(256)
void attn_kernel(const float* __restrict__ q, const float* __restrict__ k,
                 float* __restrict__ selfw, float* __restrict__ relw)
{
    int b = blockIdx.x;
    __shared__ float att[16][16];
    int tid = threadIdx.x;
    int i = tid >> 4, j = tid & 15;
    const float* qi = q + ((size_t)b * 16 + i) * 800;
    const float* kj = k + ((size_t)b * 16 + j) * 800;
    float acc = 0.f;
    for (int f = 0; f < 800; ++f) acc += qi[f] * kj[f];
    att[i][j] = acc * (1.0f / 28.284271247461902f);  // 1/sqrt(DS*P*P)
    __syncthreads();
    if (tid < 16) {
        int row = tid;
        float mx = att[row][0];
        for (int c = 1; c < 16; ++c) mx = fmaxf(mx, att[row][c]);
        float e[16]; float sum = 0.f;
        for (int c = 0; c < 16; ++c) { e[c] = __expf(att[row][c] - mx); sum += e[c]; }
        float inv = 1.f / sum;
        selfw[b * 16 + row] = e[row] * inv;
        int jj = 0;
        for (int c = 0; c < 16; ++c)
            if (c != row) relw[((size_t)b * 16 + row) * 15 + (jj++)] = e[c] * inv;
    }
}

// ---------------------------------------------------------------------------
// Pair aggregation: pred_i = selfw_i*xs_i + sum_j w_ij * relu(flag*(u_i+v_j)+b_rel)
// ---------------------------------------------------------------------------
__global__ __launch_bounds__(256)
void agg_pairs_kernel(const float* __restrict__ u, const float* __restrict__ v,
                      const float* __restrict__ xsv, const int* __restrict__ gidx,
                      const float* __restrict__ selfw, const float* __restrict__ relw,
                      const float* __restrict__ b_rel, float* __restrict__ pred)
{
    int bi = blockIdx.x;            // b*16 + i
    int b = bi >> 4, i = bi & 15;
    __shared__ int   jl[15];
    __shared__ float fl[15];
    __shared__ float wl[15];
    __shared__ float sw;
    int tid = threadIdx.x;
    if (tid < 15) {
        const int* gr = gidx + ((size_t)b * 240 + i * 15 + tid) * 3;
        jl[tid] = gr[1];
        fl[tid] = (float)gr[2];
        wl[tid] = relw[((size_t)b * 16 + i) * 15 + tid];
    }
    if (tid == 15) sw = selfw[b * 16 + i];
    __syncthreads();
    const float* ui  = u   + (size_t)bi * 1600;
    const float* xsi = xsv + (size_t)bi * 1600;
    for (int e = tid; e < 1600; e += 256) {
        float br = b_rel[e / 25];
        float ue = ui[e];
        float acc = sw * xsi[e];
        #pragma unroll
        for (int jj = 0; jj < 15; ++jj) {
            const float* vj = v + ((size_t)b * 16 + jl[jj]) * 1600;
            float t = fl[jj] * (ue + vj[e]) + br;
            acc += wl[jj] * fmaxf(t, 0.f);
        }
        pred[(size_t)bi * 1600 + e] = acc;
    }
}

// ---------------------------------------------------------------------------
// Per-batch layernorm over (N,D,P,P)=25600 elems, affine gamma/beta.
// ---------------------------------------------------------------------------
__global__ __launch_bounds__(256)
void layernorm_kernel(const float* __restrict__ a, const float* __restrict__ gamma,
                      const float* __restrict__ beta, float* __restrict__ an)
{
    int b = blockIdx.x;
    const float* ab = a + (size_t)b * 25600;
    __shared__ float r1[256], r2[256];
    __shared__ float mu_s, rstd_s;
    int tid = threadIdx.x;
    float s = 0.f, s2 = 0.f;
    for (int e = tid; e < 25600; e += 256) { float t = ab[e]; s += t; s2 += t * t; }
    r1[tid] = s; r2[tid] = s2;
    __syncthreads();
    for (int w = 128; w > 0; w >>= 1) {
        if (tid < w) { r1[tid] += r1[tid + w]; r2[tid] += r2[tid + w]; }
        __syncthreads();
    }
    if (tid == 0) {
        float mu  = r1[0] * (1.f / 25600.f);
        float var = r2[0] * (1.f / 25600.f) - mu * mu;
        mu_s = mu; rstd_s = rsqrtf(var + 1e-6f);
    }
    __syncthreads();
    float mu = mu_s, rstd = rstd_s;
    for (int e = tid; e < 25600; e += 256)
        an[(size_t)b * 25600 + e] = (ab[e] - mu) * rstd * gamma[e] + beta[e];
}

// ---------------------------------------------------------------------------
extern "C" void kernel_launch(void* const* d_in, const int* in_sizes, int n_in,
                              void* d_out, int out_size, void* d_ws, size_t ws_size,
                              hipStream_t stream)
{
    const float* x      = (const float*)d_in[0];
    const int*   gidx   = (const int*)  d_in[1];
    const float* w_rel  = (const float*)d_in[2];
    const float* b_rel  = (const float*)d_in[3];
    const float* w_self = (const float*)d_in[4];
    const float* b_self = (const float*)d_in[5];
    const float* w_aff  = (const float*)d_in[6];
    const float* b_aff  = (const float*)d_in[7];
    const float* w_agg  = (const float*)d_in[8];
    const float* b_agg  = (const float*)d_in[9];
    const float* w_attn = (const float*)d_in[10];
    const float* b_attn = (const float*)d_in[11];
    const float* w_q    = (const float*)d_in[12];
    const float* b_q    = (const float*)d_in[13];
    const float* w_k    = (const float*)d_in[14];
    const float* b_k    = (const float*)d_in[15];
    const float* gamma  = (const float*)d_in[16];
    const float* beta   = (const float*)d_in[17];
    float* out = (float*)d_out;
    (void)in_sizes; (void)n_in; (void)out_size; (void)ws_size;

    char* ws = (char*)d_ws;
    size_t off = 0;
    auto carve = [&](size_t bytes) -> char* {
        char* p = ws + off;
        off += (bytes + 255) & ~(size_t)255;
        return p;
    };
    const size_t BIGB = (size_t)1638400 * 4;          // B*N*D*P*P floats
    float* u    = (float*)carve(BIGB);
    float* v    = (float*)carve(BIGB);
    float* xs   = (float*)carve(BIGB);
    float* pred = (float*)carve(BIGB);
    float* abuf = (float*)carve(BIGB);
    float* an   = u;   // u dead after agg_pairs
    float* tmp  = v;   // v dead after agg_pairs
    float* s2d  = (float*)carve((size_t)819200 * 4);  // B*N*DS*25
    float* q    = (float*)carve((size_t)819200 * 4);
    float* k    = (float*)carve((size_t)819200 * 4);
    float* selfw= (float*)carve((size_t)1024 * 4);
    float* relw = (float*)carve((size_t)15360 * 4);
    _Float16* whRelA = (_Float16*)carve((size_t)36864 * 2);
    _Float16* whRelB = (_Float16*)carve((size_t)36864 * 2);
    _Float16* whSelf = (_Float16*)carve((size_t)36864 * 2);
    _Float16* whAff  = (_Float16*)carve((size_t)36864 * 2);
    _Float16* whAggA = (_Float16*)carve((size_t)36864 * 2);
    _Float16* whAggB = (_Float16*)carve((size_t)36864 * 2);
    _Float16* whQ    = (_Float16*)carve((size_t)9216  * 2);
    _Float16* whK    = (_Float16*)carve((size_t)9216  * 2);

    // weight repack (f32 -> f16, split concat filters along input channels)
    cvt_w_kernel<<<144, 256, 0, stream>>>(w_rel,  whRelA, 64, 128, 0,  64);
    cvt_w_kernel<<<144, 256, 0, stream>>>(w_rel,  whRelB, 64, 128, 64, 64);
    cvt_w_kernel<<<144, 256, 0, stream>>>(w_self, whSelf, 64, 64,  0,  64);
    cvt_w_kernel<<<144, 256, 0, stream>>>(w_aff,  whAff,  64, 64,  0,  64);
    cvt_w_kernel<<<144, 256, 0, stream>>>(w_agg,  whAggA, 64, 128, 0,  64);
    cvt_w_kernel<<<144, 256, 0, stream>>>(w_agg,  whAggB, 64, 128, 64, 64);
    cvt_w_kernel<<<36,  256, 0, stream>>>(w_q,    whQ,    32, 32,  0,  32);
    cvt_w_kernel<<<36,  256, 0, stream>>>(w_k,    whK,    32, 32,  0,  32);

    // attention path (tiny)
    s2d_kernel<<<3200, 256, 0, stream>>>(x, w_attn, b_attn, s2d);
    conv3x3_wmma_kernel<32,32><<<1024, 32, 0, stream>>>(s2d, whQ, b_q, nullptr, q, 1);
    conv3x3_wmma_kernel<32,32><<<1024, 32, 0, stream>>>(s2d, whK, b_k, nullptr, k, 1);
    attn_kernel<<<64, 256, 0, stream>>>(q, k, selfw, relw);

    // factored relation convs + self conv
    conv3x3_wmma_kernel<64,64><<<1024, 32, 0, stream>>>(x, whRelA, nullptr, nullptr, u, 0);
    conv3x3_wmma_kernel<64,64><<<1024, 32, 0, stream>>>(x, whRelB, nullptr, nullptr, v, 0);
    conv3x3_wmma_kernel<64,64><<<1024, 32, 0, stream>>>(x, whSelf, b_self, nullptr, xs, 1);

    // pairwise aggregation (elementwise)
    agg_pairs_kernel<<<1024, 256, 0, stream>>>(u, v, xs, gidx, selfw, relw, b_rel, pred);

    // affine conv + residual, layernorm, split-agg conv -> output
    conv3x3_wmma_kernel<64,64><<<1024, 32, 0, stream>>>(pred, whAff, b_aff, x, abuf, 1 | 4);
    layernorm_kernel<<<64, 256, 0, stream>>>(abuf, gamma, beta, an);
    conv3x3_wmma_kernel<64,64><<<1024, 32, 0, stream>>>(an, whAggA, nullptr, nullptr, tmp, 0);
    conv3x3_wmma_kernel<64,64><<<1024, 32, 0, stream>>>(x,  whAggB, b_agg,  tmp,  out, 1 | 2);
}
